// GATNet_53480932770093
// MI455X (gfx1250) — compile-verified, matrix operand order checked by reading, not run on hardware
//
#include <hip/hip_runtime.h>
#include <hip/hip_bf16.h>

// ---------------- problem constants (from reference) ----------------
#define NN   4096
#define NE   16384
#define NB   64
#define NH1  10
#define NC   512
#define ND   512
#define HC1  (NH1*NC)      // 5120
#define ETOT (NE+NN)       // 20480

#define CDIV(a,b) (((a)+(b)-1)/(b))

typedef __attribute__((ext_vector_type(16))) __bf16 v16bf;
typedef __attribute__((ext_vector_type(8)))  float  v8f;
typedef __attribute__((ext_vector_type(4)))  unsigned v4u;
typedef __attribute__((ext_vector_type(8)))  int      v8i;
typedef __attribute__((ext_vector_type(4)))  int      v4i;

#if defined(__has_builtin)
#if __has_builtin(__builtin_amdgcn_tensor_load_to_lds)
#define HAVE_TDM 1
#endif
#endif
#ifndef HAVE_TDM
#define HAVE_TDM 0
#endif

// f32 -> bf16 round-to-nearest-even via integer ops
__device__ inline __bf16 f2bf(float x) {
    unsigned u = __builtin_bit_cast(unsigned, x);
    unsigned r = (u + 0x7FFFu + ((u >> 16) & 1u)) >> 16;
    unsigned short s = (unsigned short)r;
    return __builtin_bit_cast(__bf16, s);
}

// monotonic float<->uint mapping for atomicMax on floats
__device__ inline unsigned ford(float f) {
    unsigned u = __builtin_bit_cast(unsigned, f);
    return (u & 0x80000000u) ? ~u : (u | 0x80000000u);
}
__device__ inline float funord(unsigned o) {
    unsigned u = (o & 0x80000000u) ? (o & 0x7FFFFFFFu) : ~o;
    return __builtin_bit_cast(float, u);
}
#define ORD_NEG_INF 0x007FFFFFu   // ford(-inf)

__device__ inline void edge_sd(const int* ei, int k, int* s, int* d) {
    if (k < NE) { *s = ei[k]; *d = ei[NE + k]; }
    else        { *s = *d = k - NE; }
}

#if HAVE_TDM
// Issue a 2D TDM tile load (bf16 elements) into LDS. Tile exactly covers the
// descriptor's tensor (we are always in-bounds), stride_e = global row stride.
// pad=1 adds 1 dword of LDS padding per 64B stored (matches [64][34] layout).
__device__ inline void tdm_load_2d(unsigned lds_addr, const void* gptr,
                                   unsigned tile_w, unsigned tile_h,
                                   unsigned stride_e, int pad) {
    unsigned long long ga = (unsigned long long)(size_t)gptr;
    v4u g0;
    g0[0] = 1u;                                   // count=1 (valid descriptor)
    g0[1] = lds_addr;                             // LDS byte address
    g0[2] = (unsigned)(ga & 0xFFFFFFFFu);         // global_addr[31:0]
    g0[3] = (unsigned)((ga >> 32) & 0x01FFFFFFu)  // global_addr[56:32]
          | (2u << 30);                           // type=2 ("image")
    v8i g1;
    unsigned flags = (1u << 16);                  // data_size=1 -> 2 bytes
    if (pad) flags |= (1u << 20)                  // pad_enable
                   | (3u << 22)                   // pad_interval: 16 dwords (64B)
                   | (0u << 25);                  // pad_amount: 1 dword
    g1[0] = (int)flags;                           // wg_mask=0, no barrier/iterate
    g1[1] = (int)((tile_w & 0xFFFFu) << 16);      // tensor_dim0[15:0] @ bits63:48
    g1[2] = (int)(((tile_w >> 16) & 0xFFFFu) | ((tile_h & 0xFFFFu) << 16)); // dim0 hi | dim1 lo
    g1[3] = (int)(((tile_h >> 16) & 0xFFFFu) | ((tile_w & 0xFFFFu) << 16)); // dim1 hi | tile_dim0
    g1[4] = (int)(tile_h & 0xFFFFu);              // tile_dim1 (tile_dim2=0)
    g1[5] = (int)stride_e;                        // tensor_dim0_stride[31:0]
    g1[6] = 0;                                    // stride hi, dim1_stride lo
    g1[7] = 0;
    v4i z4 = {0, 0, 0, 0};
#if __has_include(<hip/amd_detail/amd_gfx1250_TDM.h>)
    v8i z8 = {0, 0, 0, 0, 0, 0, 0, 0};
    __builtin_amdgcn_tensor_load_to_lds(g0, g1, z4, z4, z8, 0);
#else
    __builtin_amdgcn_tensor_load_to_lds(g0, g1, z4, z4, 0);
#endif
}
#endif

// ---------------- aligned bf16 WMMA GEMM, double-buffered TDM ----------------
// C = act(A[M,K]bf16 @ B[K,N]bf16 + bias); M%64==0, N%64==0, K%32==0.
// 128 threads / 4 waves, 64x64 block tile, wave = 32x32 quadrant (2x2 wmma).
// Tile t+1 is DMA'd by the TDM while tile t is consumed by the WMMAs; wave 0
// uses s_wait_tensorcnt 2 (two just-issued ops outstanding; tensor ops retire
// in order, so tile t's pair is complete).
__global__ void __launch_bounds__(128)
gemm_bf16_k(const __bf16* __restrict__ A, int lda,
            const __bf16* __restrict__ B, int ldb,
            const float* __restrict__ bias,
            float* __restrict__ Cf, __bf16* __restrict__ Cb, int ldc,
            int M, int Nn, int K, int act)
{
    __shared__ __bf16 As[2][64][34];   // 68B row stride (64B data + 4B pad)
    __shared__ __bf16 Bs[2][32][64];   // [k][n], contiguous

    const int tid  = threadIdx.x;
    const int wave = tid >> 5;
    const int lane = tid & 31;
    const int bm   = blockIdx.x * 64;
    const int bn   = blockIdx.y * 64;
    const int wm   = (wave >> 1) * 32;
    const int wn   = (wave & 1) * 32;
    const int half = lane >> 4;
    const int r16  = lane & 15;
    const int nk   = K >> 5;

    v8f acc[2][2] = {};

#if HAVE_TDM
    if (tid < 32) {   // wave 0 drives the tensor DMA: preload tile 0
        tdm_load_2d((unsigned)(size_t)(void*)&As[0][0][0],
                    A + (size_t)bm * lda, 32u, 64u, (unsigned)lda, 1);
        tdm_load_2d((unsigned)(size_t)(void*)&Bs[0][0][0],
                    B + bn, 64u, 32u, (unsigned)ldb, 0);
    }
#endif

    for (int t = 0; t < nk; ++t) {
        const int cur = t & 1;
#if HAVE_TDM
        if (tid < 32) {
            if (t + 1 < nk) {      // issue next tile into the other buffer
                const int nxt = (t + 1) & 1;
                const int k1  = (t + 1) << 5;
                tdm_load_2d((unsigned)(size_t)(void*)&As[nxt][0][0],
                            A + (size_t)bm * lda + k1, 32u, 64u, (unsigned)lda, 1);
                tdm_load_2d((unsigned)(size_t)(void*)&Bs[nxt][0][0],
                            B + (size_t)k1 * ldb + bn, 64u, 32u, (unsigned)ldb, 0);
                __builtin_amdgcn_s_wait_tensorcnt(2);   // tile t complete
            } else {
                __builtin_amdgcn_s_wait_tensorcnt(0);   // final tile complete
            }
        }
        __syncthreads();
#else
        {
            const int k0 = t << 5;
            // aligned manual staging: A rows 64B (dword stores into padded rows)
            for (int i = tid; i < 64 * 16; i += 128) {
                int r = i >> 4, seg = i & 15;
                ((unsigned*)&As[cur][r][0])[seg] =
                    ((const unsigned*)(A + (size_t)(bm + r) * lda + k0))[seg];
            }
            // B rows 128B: 16B vector copies
            for (int i = tid; i < 32 * 8; i += 128) {
                int r = i >> 3, seg = i & 7;
                ((uint4*)&Bs[cur][r][0])[seg] =
                    ((const uint4*)(B + (size_t)(k0 + r) * ldb + bn))[seg];
            }
            if (k0 + 32 < K)
                __builtin_prefetch(A + (size_t)(bm + (tid & 63)) * lda + k0 + 32, 0, 0);
        }
        __syncthreads();
#endif

        // fragments per ISA 16-bit layouts:
        // lane<16: K = {2v,2v+1} (v<4) / {16+2(v-4),...}; lane>=16: +8
        v16bf af[2], bfr[2];
        #pragma unroll
        for (int s = 0; s < 2; ++s) {
            int row = wm + s * 16 + r16;
            int col = wn + s * 16 + r16;
            #pragma unroll
            for (int vv = 0; vv < 8; ++vv) {
                int kk = ((vv < 4) ? (2 * vv) : (16 + 2 * (vv - 4))) + half * 8;
                af[s][2 * vv]      = As[cur][row][kk];
                af[s][2 * vv + 1]  = As[cur][row][kk + 1];
                bfr[s][2 * vv]     = Bs[cur][kk][col];
                bfr[s][2 * vv + 1] = Bs[cur][kk + 1][col];
            }
        }
        #pragma unroll
        for (int i = 0; i < 2; ++i)
            #pragma unroll
            for (int j = 0; j < 2; ++j)
                acc[i][j] = __builtin_amdgcn_wmma_f32_16x16x32_bf16(
                    false, af[i], false, bfr[j], (short)0, acc[i][j], false, false);
        __syncthreads();   // nobody may still read buf cur when it refills at t+2
    }

    // epilogue: D element (M = vv + 8*half, N = r16) per 16x16 tile
    #pragma unroll
    for (int i = 0; i < 2; ++i)
        #pragma unroll
        for (int j = 0; j < 2; ++j) {
            int col = bn + wn + j * 16 + r16;
            #pragma unroll
            for (int vv = 0; vv < 8; ++vv) {
                int row = bm + wm + i * 16 + vv + half * 8;
                float val = acc[i][j][vv];
                if (bias) val += bias[col];
                if (act == 1) val = val > 0.f ? val : 0.f;
                if (Cf) Cf[(size_t)row * ldc + col] = val;
                if (Cb) Cb[(size_t)row * ldc + col] = f2bf(val);
            }
        }
}

// ---------------- conversion kernels ----------------
__global__ void conv_k(const float* __restrict__ s, __bf16* d, long long n) {
    long long t = (long long)blockIdx.x * blockDim.x + threadIdx.x;
    if (t < n) d[t] = f2bf(s[t]);
}
// zero-padded convert: src[srows,scols] -> dst[drows,dcols]
__global__ void pad_conv_k(const float* __restrict__ s, __bf16* d,
                           int srows, int scols, int drows, int dcols) {
    long long t = (long long)blockIdx.x * blockDim.x + threadIdx.x;
    if (t >= (long long)drows * dcols) return;
    int r = (int)(t / dcols), c = (int)(t % dcols);
    d[t] = (r < srows && c < scols) ? f2bf(s[(size_t)r * scols + c]) : f2bf(0.f);
}
// extract v-proj slice in_w[i][:, 2D:3D) -> [2][512][512] bf16
__global__ void slice_conv_k(const float* __restrict__ s, __bf16* d) {
    long long t = (long long)blockIdx.x * blockDim.x + threadIdx.x;
    if (t >= 2LL * ND * ND) return;
    int i = (int)(t / (ND * ND));
    int r = (int)(t / ND) % ND;
    int c = (int)(t % ND);
    d[t] = f2bf(s[(size_t)i * ND * 3 * ND + (size_t)r * 3 * ND + 2 * ND + c]);
}

// ---------------- utility / graph kernels ----------------
__global__ void fill_f32_k(float* p, float v, long long n) {
    long long i = (long long)blockIdx.x * blockDim.x + threadIdx.x;
    if (i < n) p[i] = v;
}
__global__ void fill_u32_k(unsigned* p, unsigned v, long long n) {
    long long i = (long long)blockIdx.x * blockDim.x + threadIdx.x;
    if (i < n) p[i] = v;
}

__global__ void edge_accum_k(const float* __restrict__ eattr, const int* __restrict__ ei,
                             float* esum, float* cnt) {
    long long t = (long long)blockIdx.x * blockDim.x + threadIdx.x;
    if (t >= (long long)NE * 11) return;
    int e = (int)(t / 11), j = (int)(t % 11);
    int d = ei[NE + e];
    atomicAdd(&esum[(size_t)d * 11 + j], eattr[t]);
    if (j == 0) atomicAdd(&cnt[d], 1.0f);
}
__global__ void loop_norm_k(float* esum, const float* cnt) {
    int t = blockIdx.x * blockDim.x + threadIdx.x;
    if (t >= NN * 11) return;
    esum[t] = esum[t] / fmaxf(cnt[t / 11], 1.0f);
}

__global__ void __launch_bounds__(128)
rowdot_k(const float* __restrict__ xs, const float* __restrict__ as_,
         const float* __restrict__ ad_, float* slog, float* dlog, int H) {
    __shared__ float r1[128], r2[128];
    int nh = blockIdx.x;
    int h = nh % H, n = nh / H;
    const float* xr = xs + (size_t)n * H * NC + (size_t)h * NC;
    float a = 0.f, b = 0.f;
    for (int c = threadIdx.x; c < NC; c += 128) {
        float x = xr[c];
        a += x * as_[(size_t)h * NC + c];
        b += x * ad_[(size_t)h * NC + c];
    }
    r1[threadIdx.x] = a; r2[threadIdx.x] = b; __syncthreads();
    for (int s = 64; s > 0; s >>= 1) {
        if (threadIdx.x < s) { r1[threadIdx.x] += r1[threadIdx.x + s]; r2[threadIdx.x] += r2[threadIdx.x + s]; }
        __syncthreads();
    }
    if (threadIdx.x == 0) { slog[nh] = r1[0]; dlog[nh] = r2[0]; }
}

__global__ void __launch_bounds__(128)
wae_k(const float* __restrict__ We, const float* __restrict__ ae, float* wae, int H) {
    __shared__ float r[128];
    int jh = blockIdx.x;
    int h = jh % H, j = jh / H;
    float a = 0.f;
    for (int c = threadIdx.x; c < NC; c += 128)
        a += We[(size_t)j * H * NC + (size_t)h * NC + c] * ae[(size_t)h * NC + c];
    r[threadIdx.x] = a; __syncthreads();
    for (int s = 64; s > 0; s >>= 1) {
        if (threadIdx.x < s) r[threadIdx.x] += r[threadIdx.x + s];
        __syncthreads();
    }
    if (threadIdx.x == 0) wae[jh] = r[0];
}

__global__ void edge_logit_k(const int* __restrict__ ei, const float* __restrict__ eattr,
                             const float* __restrict__ lattr, const float* __restrict__ slog,
                             const float* __restrict__ dlog, const float* __restrict__ wae,
                             float* logit, unsigned* mbuf, int H) {
    long long t = (long long)blockIdx.x * blockDim.x + threadIdx.x;
    if (t >= (long long)ETOT * H) return;
    int k = (int)(t / H), h = (int)(t % H);
    int s, d; edge_sd(ei, k, &s, &d);
    const float* row = (k < NE) ? (eattr + (size_t)k * 11) : (lattr + (size_t)(k - NE) * 11);
    float ee = 0.f;
    #pragma unroll
    for (int j = 0; j < 11; ++j) ee += row[j] * wae[j * H + h];
    float lg = slog[(size_t)s * H + h] + dlog[(size_t)d * H + h] + ee;
    lg = lg > 0.f ? lg : 0.2f * lg;
    logit[t] = lg;
    atomicMax(&mbuf[(size_t)d * H + h], ford(lg));
}

__global__ void edge_exp_k(const int* __restrict__ ei, float* __restrict__ logit,
                           const unsigned* __restrict__ mbuf, float* psum, int H) {
    long long t = (long long)blockIdx.x * blockDim.x + threadIdx.x;
    if (t >= (long long)ETOT * H) return;
    int k = (int)(t / H), h = (int)(t % H);
    int s, d; edge_sd(ei, k, &s, &d); (void)s;
    float p = expf(logit[t] - funord(mbuf[(size_t)d * H + h]));
    logit[t] = p;
    atomicAdd(&psum[(size_t)d * H + h], p);
}

__global__ void init_bias_k(float* out, const float* __restrict__ bias, int HC) {
    long long t = (long long)blockIdx.x * blockDim.x + threadIdx.x;
    if (t >= (long long)NN * HC) return;
    out[t] = bias[t % HC];
}

__global__ void gat_agg_k(const int* __restrict__ ei, const float* __restrict__ xs,
                          const float* __restrict__ p, const float* __restrict__ psum,
                          float* out, int H, int HC) {
    long long t = (long long)blockIdx.x * blockDim.x + threadIdx.x;
    if (t >= (long long)ETOT * HC) return;
    int k = (int)(t / HC), ch = (int)(t % HC);
    int h = ch >> 9;   // C == 512
    int s, d; edge_sd(ei, k, &s, &d);
    float w = p[(size_t)k * H + h] / psum[(size_t)d * H + h];
    atomicAdd(&out[(size_t)d * HC + ch], xs[(size_t)s * HC + ch] * w);
}

// ELU in place (f32) + bf16 copy for the following GEMM
__global__ void elu_dual_k(float* x, __bf16* xb, long long n) {
    long long t = (long long)blockIdx.x * blockDim.x + threadIdx.x;
    if (t >= n) return;
    float v = x[t];
    v = v > 0.f ? v : (expf(v) - 1.f);
    x[t] = v;
    xb[t] = f2bf(v);
}

__global__ void alpha_out_k(const int* __restrict__ ei, const float* __restrict__ p,
                            const float* __restrict__ psum, float* out) {
    int k = blockIdx.x * blockDim.x + threadIdx.x;
    if (k >= ETOT) return;
    int s, d; edge_sd(ei, k, &s, &d); (void)s;
    out[k] = p[k] / psum[d];
}

// LayerNorm(x + res)*g + b, dual f32 + bf16 outputs
__global__ void __launch_bounds__(256)
ln_k(const float* __restrict__ x, const float* __restrict__ res,
     const float* __restrict__ g, const float* __restrict__ b,
     float* out, __bf16* outb) {
    __shared__ float s1[256], s2[256];
    int row = blockIdx.x, tid = threadIdx.x;
    const float* xr = x + (size_t)row * ND;
    const float* rr = res + (size_t)row * ND;
    float a1 = 0.f, a2 = 0.f;
    for (int c = tid; c < ND; c += 256) { float v = xr[c] + rr[c]; a1 += v; a2 += v * v; }
    s1[tid] = a1; s2[tid] = a2; __syncthreads();
    for (int s = 128; s > 0; s >>= 1) {
        if (tid < s) { s1[tid] += s1[tid + s]; s2[tid] += s2[tid + s]; }
        __syncthreads();
    }
    float mu = s1[0] * (1.f / ND);
    float var = s2[0] * (1.f / ND) - mu * mu;
    float inv = rsqrtf(var + 1e-5f);
    for (int c = tid; c < ND; c += 256) {
        float v = (xr[c] + rr[c] - mu) * inv * g[c] + b[c];
        out[(size_t)row * ND + c] = v;
        outb[(size_t)row * ND + c] = f2bf(v);
    }
}

__global__ void pool_max_k(const float* __restrict__ h, const int* __restrict__ batch,
                           unsigned* out) {
    long long t = (long long)blockIdx.x * blockDim.x + threadIdx.x;
    if (t >= (long long)NN * ND) return;
    int n = (int)(t / ND), c = (int)(t % ND);
    atomicMax(&out[(size_t)batch[n] * ND + c], ford(h[t]));
}
__global__ void pool_fix_k(unsigned* out) {
    int t = blockIdx.x * blockDim.x + threadIdx.x;
    if (t >= NB * ND) return;
    out[t] = __builtin_bit_cast(unsigned, funord(out[t]));
}

// ---------------- host orchestration ----------------
extern "C" void kernel_launch(void* const* d_in, const int* in_sizes, int n_in,
                              void* d_out, int out_size, void* d_ws, size_t ws_size,
                              hipStream_t stream) {
    (void)in_sizes; (void)n_in; (void)out_size; (void)ws_size;

    const float* x1    = (const float*)d_in[0];
    const int*   ei    = (const int*)d_in[1];
    const float* eattr = (const float*)d_in[2];
    const int*   batch = (const int*)d_in[3];
    const float* W1    = (const float*)d_in[4];
    const float* as1   = (const float*)d_in[5];
    const float* ad1   = (const float*)d_in[6];
    const float* We1   = (const float*)d_in[7];
    const float* ae1   = (const float*)d_in[8];
    const float* b1    = (const float*)d_in[9];
    const float* W2    = (const float*)d_in[10];
    const float* as2   = (const float*)d_in[11];
    const float* ad2   = (const float*)d_in[12];
    const float* We2   = (const float*)d_in[13];
    const float* ae2   = (const float*)d_in[14];
    const float* b2    = (const float*)d_in[15];
    const float* in_w  = (const float*)d_in[16];
    const float* in_b  = (const float*)d_in[17];
    const float* out_w = (const float*)d_in[18];
    const float* out_b = (const float*)d_in[19];
    const float* ln1g  = (const float*)d_in[20];
    const float* ln1b  = (const float*)d_in[21];
    const float* f1w   = (const float*)d_in[22];
    const float* f1b   = (const float*)d_in[23];
    const float* f2w   = (const float*)d_in[24];
    const float* f2b   = (const float*)d_in[25];
    const float* ln2g  = (const float*)d_in[26];
    const float* ln2b  = (const float*)d_in[27];

    float* dout = (float*)d_out;

    // workspace carve-up
    char* wsb = (char*)d_ws;
    size_t off = 0;
    auto allocf = [&](size_t n) { float* p = (float*)(wsb + off); off += ((n * 4 + 255) & ~(size_t)255); return p; };
    auto allocb = [&](size_t n) { __bf16* p = (__bf16*)(wsb + off); off += ((n * 2 + 255) & ~(size_t)255); return p; };

    float* lattr = allocf((size_t)NN * 11);
    float* cnt   = allocf(NN);
    float* xs1   = allocf((size_t)NN * HC1);
    float* slog1 = allocf((size_t)NN * NH1);
    float* dlog1 = allocf((size_t)NN * NH1);
    float* wae1  = allocf(11 * NH1);
    float* m1    = allocf((size_t)NN * NH1);
    float* p1    = allocf((size_t)ETOT * NH1);
    float* ps1   = allocf((size_t)NN * NH1);
    float* h1    = allocf((size_t)NN * HC1);
    float* xs2   = allocf((size_t)NN * NC);
    float* slog2 = allocf(NN);
    float* dlog2 = allocf(NN);
    float* wae2  = allocf(16);
    float* m2    = allocf(NN);
    float* p2    = allocf(ETOT);
    float* ps2   = allocf(NN);
    float* hA    = allocf((size_t)NN * ND);
    float* hB    = allocf((size_t)NN * ND);
    float* ab    = allocf((size_t)NN * ND);

    __bf16* x1b  = allocb((size_t)NN * 96);
    __bf16* W1b  = allocb((size_t)96 * HC1);
    __bf16* W2b  = allocb((size_t)HC1 * NC);
    __bf16* inwb = allocb(2LL * ND * ND);
    __bf16* outwb= allocb(2LL * ND * ND);
    __bf16* f1wb = allocb(2LL * ND * 4 * ND);
    __bf16* f2wb = allocb(2LL * 4 * ND * ND);
    __bf16* h1b  = allocb((size_t)NN * HC1);
    __bf16* hAb  = allocb((size_t)NN * ND);
    __bf16* hBb  = allocb((size_t)NN * ND);
    __bf16* vbb  = allocb((size_t)NN * ND);
    __bf16* fbb  = allocb((size_t)NN * 4 * ND);

    auto gemm = [&](const __bf16* A, int lda, const __bf16* Bm, int ldb,
                    const float* bias, float* Cf, __bf16* Cb, int ldc,
                    int M, int Nn, int K, int act) {
        dim3 g(CDIV(M, 64), CDIV(Nn, 64));
        gemm_bf16_k<<<g, 128, 0, stream>>>(A, lda, Bm, ldb, bias, Cf, Cb, ldc, M, Nn, K, act);
    };

    // ---- weight / input conversion to bf16 (padded where ragged) ----
    pad_conv_k<<<CDIV(NN * 96, 256), 256, 0, stream>>>(x1, x1b, NN, 93, NN, 96);
    pad_conv_k<<<CDIV(96 * HC1, 256), 256, 0, stream>>>(W1, W1b, 93, HC1, 96, HC1);
    conv_k<<<CDIV((long long)HC1 * NC, 256), 256, 0, stream>>>(W2, W2b, (long long)HC1 * NC);
    slice_conv_k<<<CDIV(2 * ND * ND, 256), 256, 0, stream>>>(in_w, inwb);
    conv_k<<<CDIV(2LL * ND * ND, 256), 256, 0, stream>>>(out_w, outwb, 2LL * ND * ND);
    conv_k<<<CDIV(2LL * ND * 4 * ND, 256), 256, 0, stream>>>(f1w, f1wb, 2LL * ND * 4 * ND);
    conv_k<<<CDIV(2LL * 4 * ND * ND, 256), 256, 0, stream>>>(f2w, f2wb, 2LL * 4 * ND * ND);

    // ---- self-loop edge attributes (mean of incoming) ----
    fill_f32_k<<<CDIV(NN * 11, 256), 256, 0, stream>>>(lattr, 0.f, (long long)NN * 11);
    fill_f32_k<<<CDIV(NN, 256), 256, 0, stream>>>(cnt, 0.f, NN);
    edge_accum_k<<<CDIV(NE * 11, 256), 256, 0, stream>>>(eattr, ei, lattr, cnt);
    loop_norm_k<<<CDIV(NN * 11, 256), 256, 0, stream>>>(lattr, cnt);

    // ---- GAT layer 1 (10 heads x 512) ----
    gemm(x1b, 96, W1b, HC1, nullptr, xs1, nullptr, HC1, NN, HC1, 96, 0);
    rowdot_k<<<NN * NH1, 128, 0, stream>>>(xs1, as1, ad1, slog1, dlog1, NH1);
    wae_k<<<11 * NH1, 128, 0, stream>>>(We1, ae1, wae1, NH1);
    fill_u32_k<<<CDIV(NN * NH1, 256), 256, 0, stream>>>((unsigned*)m1, ORD_NEG_INF, (long long)NN * NH1);
    fill_f32_k<<<CDIV(NN * NH1, 256), 256, 0, stream>>>(ps1, 0.f, (long long)NN * NH1);
    edge_logit_k<<<CDIV(ETOT * NH1, 256), 256, 0, stream>>>(ei, eattr, lattr, slog1, dlog1, wae1, p1, (unsigned*)m1, NH1);
    edge_exp_k<<<CDIV(ETOT * NH1, 256), 256, 0, stream>>>(ei, p1, (unsigned*)m1, ps1, NH1);
    init_bias_k<<<CDIV((long long)NN * HC1, 256), 256, 0, stream>>>(h1, b1, HC1);
    {
        long long tot = (long long)ETOT * HC1;
        gat_agg_k<<<(unsigned)CDIV(tot, 256), 256, 0, stream>>>(ei, xs1, p1, ps1, h1, NH1, HC1);
    }
    elu_dual_k<<<CDIV((long long)NN * HC1, 256), 256, 0, stream>>>(h1, h1b, (long long)NN * HC1);

    // ---- GAT layer 2 (1 head x 512) ----
    gemm(h1b, HC1, W2b, NC, nullptr, xs2, nullptr, NC, NN, NC, HC1, 0);
    rowdot_k<<<NN, 128, 0, stream>>>(xs2, as2, ad2, slog2, dlog2, 1);
    wae_k<<<11, 128, 0, stream>>>(We2, ae2, wae2, 1);
    fill_u32_k<<<CDIV(NN, 256), 256, 0, stream>>>((unsigned*)m2, ORD_NEG_INF, NN);
    fill_f32_k<<<CDIV(NN, 256), 256, 0, stream>>>(ps2, 0.f, NN);
    edge_logit_k<<<CDIV(ETOT, 256), 256, 0, stream>>>(ei, eattr, lattr, slog2, dlog2, wae2, p2, (unsigned*)m2, 1);
    edge_exp_k<<<CDIV(ETOT, 256), 256, 0, stream>>>(ei, p2, (unsigned*)m2, ps2, 1);
    alpha_out_k<<<CDIV(ETOT, 256), 256, 0, stream>>>(ei, p2, ps2, dout + (size_t)NB * ND);
    init_bias_k<<<CDIV((long long)NN * NC, 256), 256, 0, stream>>>(hA, b2, NC);
    {
        long long tot = (long long)ETOT * NC;
        gat_agg_k<<<(unsigned)CDIV(tot, 256), 256, 0, stream>>>(ei, xs2, p2, ps2, hA, 1, NC);
    }
    elu_dual_k<<<CDIV((long long)NN * ND, 256), 256, 0, stream>>>(hA, hAb, (long long)NN * ND);

    // ---- Transformer encoder x2 (seq_len=1: attn == out_proj(v_proj(x))) ----
    for (int i = 0; i < 2; ++i) {
        const float* vB  = in_b + (size_t)i * 3 * ND + 2 * ND;
        const float* oB  = out_b + (size_t)i * ND;
        const float* g1  = ln1g + (size_t)i * ND;  const float* bb1 = ln1b + (size_t)i * ND;
        const float* b1f = f1b + (size_t)i * 4 * ND;
        const float* b2f = f2b + (size_t)i * ND;
        const float* g2  = ln2g + (size_t)i * ND;  const float* bb2 = ln2b + (size_t)i * ND;

        gemm(hAb, ND, inwb + (size_t)i * ND * ND, ND, vB, nullptr, vbb, ND, NN, ND, ND, 0);
        gemm(vbb, ND, outwb + (size_t)i * ND * ND, ND, oB, ab, nullptr, ND, NN, ND, ND, 0);
        ln_k<<<NN, 256, 0, stream>>>(hA, ab, g1, bb1, hB, hBb);
        gemm(hBb, ND, f1wb + (size_t)i * ND * 4 * ND, 4 * ND, b1f, nullptr, fbb, 4 * ND, NN, 4 * ND, ND, 1);
        gemm(fbb, 4 * ND, f2wb + (size_t)i * 4 * ND * ND, ND, b2f, ab, nullptr, ND, NN, ND, 4 * ND, 0);
        ln_k<<<NN, 256, 0, stream>>>(hB, ab, g2, bb2, hA, hAb);
    }

    // ---- global max pool into d_out[0 : 64*512) ----
    fill_u32_k<<<CDIV(NB * ND, 256), 256, 0, stream>>>((unsigned*)dout, ORD_NEG_INF, (long long)NB * ND);
    pool_max_k<<<CDIV((long long)NN * ND, 256), 256, 0, stream>>>(hA, batch, (unsigned*)dout);
    pool_fix_k<<<CDIV(NB * ND, 256), 256, 0, stream>>>((unsigned*)dout);
}
